// LSHLayer_25537875542392
// MI455X (gfx1250) — matrix operand order repaired and checked by reading, not compiled, not run on hardware
//
#include <hip/hip_runtime.h>
#include <hip/hip_bf16.h>
#include <float.h>

// Problem constants (from reference)
#define BSZ   256
#define DDIM  1024
#define CDIM  262144
#define SDIM  32768

typedef __attribute__((ext_vector_type(16))) __bf16 v16bf;
typedef __attribute__((ext_vector_type(8)))  __bf16 v8bf;
typedef __attribute__((ext_vector_type(4)))  __bf16 v4bf;
typedef __attribute__((ext_vector_type(8)))  float  v8f;

// LDS B-stripe: 64 columns x 32 k-values (bf16) per buffer, padded to 80B/col
// (40 elements) so 16-lane fragment reads hit distinct banks. Two buffers.
#define BCOL_ELEMS 40           // 32 data + 8 pad (bf16 elements), 80 bytes
#define BBUF_ELEMS (64 * BCOL_ELEMS)

// ---------------------------------------------------------------------------
// Kernel 1: gather sampled weight rows and convert fp32 -> bf16, gather bias.
// ---------------------------------------------------------------------------
__global__ __launch_bounds__(256) void k_gather_convert(
    const float* __restrict__ weight, const float* __restrict__ bias,
    const int* __restrict__ ids, __bf16* __restrict__ wb,
    float* __restrict__ sbias)
{
    const int row = blockIdx.x;            // 0..SDIM-1
    const int id  = ids[row];
    const float4* src = (const float4*)(weight + (size_t)id * DDIM);
    v4bf*         dst = (v4bf*)(wb + (size_t)row * DDIM);
    const int t = threadIdx.x;
    float4 v = src[t];
    v4bf o;
    o[0] = (__bf16)v.x; o[1] = (__bf16)v.y; o[2] = (__bf16)v.z; o[3] = (__bf16)v.w;
    dst[t] = o;
    if (t == 0) sbias[row] = bias[id];
}

// ---------------------------------------------------------------------------
// Kernel 2: convert x fp32 -> bf16 (256x1024).
// ---------------------------------------------------------------------------
__global__ __launch_bounds__(256) void k_convert_x(
    const float* __restrict__ x, __bf16* __restrict__ xb)
{
    const int i = blockIdx.x * blockDim.x + threadIdx.x;   // over BSZ*DDIM/4
    float4 v = ((const float4*)x)[i];
    v4bf o;
    o[0] = (__bf16)v.x; o[1] = (__bf16)v.y; o[2] = (__bf16)v.z; o[3] = (__bf16)v.w;
    ((v4bf*)xb)[i] = o;
}

// ---------------------------------------------------------------------------
// Kernel 3: logits GEMM with v_wmma_f32_16x16x32_bf16.
// Block = 256 threads = 8 waves -> 128 rows x 64 cols tile.
// All 8 waves share the B stripe, double-buffered in LDS and filled with
// async global->LDS copies (ASYNCcnt) one k-step ahead of compute.
// Fragment layouts used by this kernel (A/B in contiguous K order):
//   A 16x32 bf16 : lane l (M=l&15) holds 16 contiguous K at 16*(l>>4)
//   B 32x16 bf16 : lane l (N=l&15) holds 16 contiguous K at 16*(l>>4)
//   C/D 16x16 f32: lane l covers N=l&15; VGPR r -> M = r + 8*(l>>4)
// ---------------------------------------------------------------------------
__global__ __launch_bounds__(256) void k_gemm_wmma(
    const __bf16* __restrict__ xb, const __bf16* __restrict__ wb,
    const float* __restrict__ sbias, float* __restrict__ logits)
{
    __shared__ __align__(16) __bf16 bsh[2][BBUF_ELEMS];

    const int t      = threadIdx.x;
    const int waveId = t >> 5;
    const int lane   = t & 31;
    const int mr     = lane & 15;            // M (A) / N (B,C,D)
    const int kh     = lane >> 4;            // half-wave selector

    const int nColTiles = SDIM / 64;         // 512
    const int blockRow  = blockIdx.x / nColTiles;     // 0..1
    const int colTile   = blockIdx.x % nColTiles;
    const int m0 = (blockRow * 8 + waveId) * 16;      // wave's 16-row slice
    const int c0 = colTile * 64;

    // Cooperative async fill: thread t copies 16B: col = t/4, chunk = t%4.
    const int fcol = t >> 2;
    const int fchk = t & 3;
    const __bf16* gB = wb + (size_t)(c0 + fcol) * DDIM + fchk * 8;
    const unsigned ldsDst0 =
        (unsigned)(uintptr_t)&bsh[0][fcol * BCOL_ELEMS + fchk * 8];
    const unsigned ldsDst1 =
        (unsigned)(uintptr_t)&bsh[1][fcol * BCOL_ELEMS + fchk * 8];

    v8f acc0 = {}; v8f acc1 = {}; v8f acc2 = {}; v8f acc3 = {};

    const __bf16* aRow = xb + (size_t)(m0 + mr) * DDIM;

    // Prologue: stage k0=0 stripe into buffer 0.
    asm volatile("global_load_async_to_lds_b128 %0, %1, off"
                 :: "v"(ldsDst0), "v"(gB) : "memory");
    asm volatile("s_wait_asynccnt 0x0" ::: "memory");
    __syncthreads();

    int p = 0;
    for (int k0 = 0; k0 < DDIM; k0 += 32) {
        // Kick off next stripe into the other buffer (overlaps with WMMAs).
        if (k0 + 32 < DDIM) {
            asm volatile("global_load_async_to_lds_b128 %0, %1, off"
                         :: "v"(p ? ldsDst0 : ldsDst1), "v"(gB + k0 + 32)
                         : "memory");
        }

        // A fragment straight from global (x is tiny, L2-resident).
        union { v16bf v; v8bf h[2]; } a;
        a.h[0] = *(const v8bf*)(aRow + k0 + 16 * kh);
        a.h[1] = *(const v8bf*)(aRow + k0 + 16 * kh + 8);

        // B fragments from LDS (staged one iteration ago).
        const __bf16* bb = &bsh[p][mr * BCOL_ELEMS + kh * 16];
        v16bf b0 = *(const v16bf*)(bb);
        v16bf b1 = *(const v16bf*)(bb + 16 * BCOL_ELEMS);
        v16bf b2 = *(const v16bf*)(bb + 32 * BCOL_ELEMS);
        v16bf b3 = *(const v16bf*)(bb + 48 * BCOL_ELEMS);

        acc0 = __builtin_amdgcn_wmma_f32_16x16x32_bf16(false, a.v, false, b0,
                                                       (short)0, acc0, false, false);
        acc1 = __builtin_amdgcn_wmma_f32_16x16x32_bf16(false, a.v, false, b1,
                                                       (short)0, acc1, false, false);
        acc2 = __builtin_amdgcn_wmma_f32_16x16x32_bf16(false, a.v, false, b2,
                                                       (short)0, acc2, false, false);
        acc3 = __builtin_amdgcn_wmma_f32_16x16x32_bf16(false, a.v, false, b3,
                                                       (short)0, acc3, false, false);

        // Own async copy must have landed; barrier flips the double buffer.
        asm volatile("s_wait_asynccnt 0x0" ::: "memory");
        __syncthreads();
        p ^= 1;
    }

    const float bv0 = sbias[c0 + mr];
    const float bv1 = sbias[c0 + 16 + mr];
    const float bv2 = sbias[c0 + 32 + mr];
    const float bv3 = sbias[c0 + 48 + mr];

    #pragma unroll
    for (int r = 0; r < 8; ++r) {
        size_t o = (size_t)(m0 + 8 * kh + r) * SDIM + c0 + mr;
        logits[o]      = acc0[r] + bv0;
        logits[o + 16] = acc1[r] + bv1;
        logits[o + 32] = acc2[r] + bv2;
        logits[o + 48] = acc3[r] + bv3;
    }
}

// ---------------------------------------------------------------------------
// Kernel 4: per-row masked log-softmax CE, online softmax, vectorized x4.
// loss_b = m + log(Z) - sum(targets * logits)   (targets sum to 1 per row)
// ---------------------------------------------------------------------------
__global__ __launch_bounds__(256) void k_row_loss(
    const float* __restrict__ logits, const float* __restrict__ tgt,
    const unsigned char* __restrict__ san, float* __restrict__ rloss)
{
    __shared__ float sm[256], sz[256], st[256];
    const int b = blockIdx.x, t = threadIdx.x;
    const float*    lrow = logits + (size_t)b * SDIM;
    const float*    trow = tgt    + (size_t)b * SDIM;
    const unsigned* srow = (const unsigned*)(san + (size_t)b * SDIM);

    float m = -FLT_MAX, z = 0.f, tl = 0.f;
    for (int s4 = t; s4 < SDIM / 4; s4 += 256) {
        const unsigned msk = srow[s4];
        const float4 l4 = ((const float4*)lrow)[s4];
        const float4 t4 = ((const float4*)trow)[s4];
        #define CE_STEP(LV, TV, SH)                                   \
            if (msk & (0xffu << SH)) {                                \
                float l = (LV);                                       \
                tl = fmaf((TV), l, tl);                               \
                if (l > m) { z *= __expf(m - l); m = l; }             \
                z += __expf(l - m);                                   \
            }
        CE_STEP(l4.x, t4.x, 0)
        CE_STEP(l4.y, t4.y, 8)
        CE_STEP(l4.z, t4.z, 16)
        CE_STEP(l4.w, t4.w, 24)
        #undef CE_STEP
    }
    sm[t] = m; sz[t] = z; st[t] = tl;
    __syncthreads();
    for (int off = 128; off > 0; off >>= 1) {
        if (t < off) {
            float m2 = sm[t + off], z2 = sz[t + off];
            float M  = fmaxf(sm[t], m2);
            float zn = 0.f;
            if (sz[t] > 0.f) zn += sz[t] * __expf(sm[t] - M);
            if (z2    > 0.f) zn += z2    * __expf(m2    - M);
            sm[t] = M; sz[t] = zn; st[t] += st[t + off];
        }
        __syncthreads();
    }
    if (t == 0) rloss[b] = sm[0] + __logf(sz[0]) - st[0];
}

// ---------------------------------------------------------------------------
// Kernel 5: mean over BSZ rows -> scalar loss.
// ---------------------------------------------------------------------------
__global__ __launch_bounds__(256) void k_mean(const float* __restrict__ rloss,
                                              float* __restrict__ out)
{
    __shared__ float s[256];
    const int t = threadIdx.x;
    s[t] = rloss[t];
    __syncthreads();
    for (int off = 128; off > 0; off >>= 1) {
        if (t < off) s[t] += s[t + off];
        __syncthreads();
    }
    if (t == 0) out[0] = s[0] * (1.0f / (float)BSZ);
}

// ---------------------------------------------------------------------------
extern "C" void kernel_launch(void* const* d_in, const int* in_sizes, int n_in,
                              void* d_out, int out_size, void* d_ws, size_t ws_size,
                              hipStream_t stream)
{
    const float* x       = (const float*)d_in[0];          // [BSZ, DDIM]
    const float* weight  = (const float*)d_in[1];          // [CDIM, DDIM]
    const float* bias    = (const float*)d_in[2];          // [CDIM]
    const float* targets = (const float*)d_in[3];          // [BSZ, SDIM]
    const int*   ids     = (const int*)d_in[4];            // [SDIM]
    const unsigned char* san = (const unsigned char*)d_in[5]; // [BSZ, SDIM] bool
    float* out = (float*)d_out;

    // workspace layout
    char* ws = (char*)d_ws;
    __bf16* wb    = (__bf16*)(ws);                                   // S*D*2   = 64 MiB
    __bf16* xb    = (__bf16*)(ws + (size_t)SDIM * DDIM * 2);         // BS*D*2  = 512 KiB
    float*  sbias = (float*) (ws + (size_t)SDIM * DDIM * 2
                                 + (size_t)BSZ  * DDIM * 2);         // S*4     = 128 KiB
    float*  logits= (float*) ((char*)sbias + (size_t)SDIM * 4);      // BS*S*4  = 32 MiB
    float*  rloss = (float*) ((char*)logits + (size_t)BSZ * SDIM * 4); // BS*4

    // 1) gather + fp32->bf16 convert of sampled weight rows (+ bias gather)
    k_gather_convert<<<SDIM, 256, 0, stream>>>(weight, bias, ids, wb, sbias);

    // 2) x -> bf16
    k_convert_x<<<(BSZ * DDIM / 4) / 256, 256, 0, stream>>>(x, xb);

    // 3) WMMA GEMM: 128x64 tile per block (8 waves), async-LDS staged B.
    {
        const int nBlocks = (BSZ / 128) * (SDIM / 64);     // 2 * 512 = 1024
        k_gemm_wmma<<<nBlocks, 256, 0, stream>>>(xb, wb, sbias, logits);
    }

    // 4) per-row masked log-softmax CE
    k_row_loss<<<BSZ, 256, 0, stream>>>(logits, targets, san, rloss);

    // 5) mean -> scalar
    k_mean<<<1, 256, 0, stream>>>(rloss, out);
}
